// Damping_72799695667307
// MI455X (gfx1250) — matrix operand order repaired and checked by reading, not compiled, not run on hardware
//
#include <hip/hip_runtime.h>
#include <math.h>

typedef unsigned short ushort_t;
typedef unsigned int   uint_t;
typedef __attribute__((ext_vector_type(16))) __bf16    v16bf;
typedef __attribute__((ext_vector_type(8)))  float     v8f;
typedef __attribute__((ext_vector_type(8)))  ushort_t  us8;
typedef __attribute__((ext_vector_type(16))) ushort_t  us16;

// ---------------- geometry ----------------
#define NB   64        // N
#define HH   256       // H
#define OFFD 2016      // N(N-1)/2
#define BM   16        // batch rows per block
#define NT   256       // threads per block (8 waves)

// LDS strides (element units), chosen for 16B alignment + bank stagger
#define ZP   2020      // f32 per z row
#define HP   264       // bf16 per hidden row
#define XP16 72        // bf16 per x row
#define XPF  68        // f32 per x/xd/y row

// LDS byte offsets (z region time-aliased with h1; y aliases h2d)
#define OFF_Z    0
#define SZ_Z     (BM*ZP*4)              /* 129280 */
#define OFF_H1D  0
#define OFF_H1O  (BM*HP*2)              /* 8448   */
#define OFF_H2D  SZ_Z                   /* 129280 */
#define OFF_H2O  (SZ_Z + BM*HP*2)       /* 137728 */
#define OFF_Y    SZ_Z                   /* alias h2d */
#define OFF_XBF  (SZ_Z + 2*BM*HP*2)     /* 146176 */
#define OFF_X0   (OFF_XBF + BM*XP16*2)  /* 148480 */
#define OFF_XD   (OFF_X0 + BM*XPF*4)    /* 152832 */
#define SMEM_TOTAL (OFF_XD + BM*XPF*4)  /* 157184 */

// bf16 weight offsets inside d_ws (ushort units)
#define WOFF_D1 0
#define WOFF_D2 16384
#define WOFF_DO 81920
#define WOFF_O1 98304
#define WOFF_O2 114688
#define WOFF_OO 180224

// native CDNA5 tanh (V_TANH_F32) with compile-safe fallbacks
#if __has_builtin(__builtin_amdgcn_tanhf)
__device__ __forceinline__ float fast_tanh(float x) { return __builtin_amdgcn_tanhf(x); }
#elif __has_builtin(__builtin_amdgcn_tanh_f32)
__device__ __forceinline__ float fast_tanh(float x) { return __builtin_amdgcn_tanh_f32(x); }
#else
__device__ __forceinline__ float fast_tanh(float x) { return tanhf(x); }
#endif

__device__ __forceinline__ ushort_t f2bf(float f) {
  uint_t u = __builtin_bit_cast(uint_t, f);
  uint_t r = (u + 0x7FFFu + ((u >> 16) & 1u)) >> 16;   // RNE
  return (ushort_t)r;
}

__global__ void wconv_kernel(const float* __restrict__ src,
                             ushort_t* __restrict__ dst, int n) {
  int i = blockIdx.x * blockDim.x + threadIdx.x;
  if (i < n) dst[i] = f2bf(src[i]);
}

// One 16x16 output tile of  A[16,K](LDS bf16) @ W[nout,K]^T(global bf16), f32 acc.
__device__ __forceinline__ v8f gemm_tile(const ushort_t* __restrict__ A, int lda,
                                         const ushort_t* __restrict__ W, int K,
                                         int n0, int nl, int hi) {
  v8f c = {};
  const ushort_t* ap = A + nl * lda + hi * 8;           // row m = nl
  const ushort_t* wp = W + (n0 + nl) * K + hi * 8;      // row n = n0+nl of W
  #pragma unroll
  for (int k0 = 0; k0 < K; k0 += 32) {
    us8 a0 = *(const us8*)(ap + k0);        // K = k0+8hi .. +7   (elems 0..7)
    us8 a1 = *(const us8*)(ap + k0 + 16);   // K = k0+8hi+16 ..   (elems 8..15)
    us8 b0 = *(const us8*)(wp + k0);        // K = k0+8hi .. +7
    us8 b1 = *(const us8*)(wp + k0 + 16);   // K = k0+8hi+16 ..
    us16 av = __builtin_shufflevector(a0, a1, 0,1,2,3,4,5,6,7,8,9,10,11,12,13,14,15);
    us16 bv = __builtin_shufflevector(b0, b1, 0,1,2,3,4,5,6,7,8,9,10,11,12,13,14,15);
    c = __builtin_amdgcn_wmma_f32_16x16x32_bf16(
        false, __builtin_bit_cast(v16bf, av),
        false, __builtin_bit_cast(v16bf, bv),
        (short)0, c, false, false);
  }
  return c;
}

__global__ __launch_bounds__(NT) void damping_fused_kernel(
    const float* __restrict__ x,
    const ushort_t* __restrict__ wbf,
    const float* __restrict__ bd1, const float* __restrict__ bd2,
    const float* __restrict__ bdo, const float* __restrict__ bo1,
    const float* __restrict__ bo2, const float* __restrict__ boo,
    float* __restrict__ out) {
  extern __shared__ char smem[];
  float*    zb  = (float*)   (smem + OFF_Z);
  ushort_t* h1d = (ushort_t*)(smem + OFF_H1D);
  ushort_t* h1o = (ushort_t*)(smem + OFF_H1O);
  ushort_t* h2d = (ushort_t*)(smem + OFF_H2D);
  ushort_t* h2o = (ushort_t*)(smem + OFF_H2O);
  float*    yb  = (float*)   (smem + OFF_Y);
  ushort_t* xbf = (ushort_t*)(smem + OFF_XBF);
  float*    x0  = (float*)   (smem + OFF_X0);
  float*    xd  = (float*)   (smem + OFF_XD);

  const int tid  = threadIdx.x;
  const int lane = tid & 31;
  const int wid  = tid >> 5;
  const int nl   = lane & 15;
  const int hi   = lane >> 4;

  // ---- load x tile: 16x64 f32, one float4 per thread ----
  {
    const float4 v = ((const float4*)(x + (size_t)blockIdx.x * BM * NB))[tid];
    int s = tid >> 4, c0 = (tid & 15) << 2;
    float* xr = x0 + s * XPF + c0;
    xr[0] = v.x; xr[1] = v.y; xr[2] = v.z; xr[3] = v.w;
    ushort_t* xb = xbf + s * XP16 + c0;
    xb[0] = f2bf(v.x); xb[1] = f2bf(v.y); xb[2] = f2bf(v.z); xb[3] = f2bf(v.w);
  }
  __syncthreads();

  // ---- layer 1: [16,64] x W1^T -> tanh -> h1, K=64 (branch-uniform loops) ----
  for (int t = wid; t < 16; t += 8) {                 // diag branch: 2 tiles/wave
    int n0 = t << 4;
    v8f c = gemm_tile(xbf, XP16, wbf + WOFF_D1, NB, n0, nl, hi);
    float bv = bd1[n0 + nl];
    #pragma unroll
    for (int r = 0; r < 8; r++)
      h1d[(r + 8 * hi) * HP + n0 + nl] = f2bf(fast_tanh(c[r] + bv));
  }
  for (int t = wid; t < 16; t += 8) {                 // offdiag branch
    int n0 = t << 4;
    v8f c = gemm_tile(xbf, XP16, wbf + WOFF_O1, NB, n0, nl, hi);
    float bv = bo1[n0 + nl];
    #pragma unroll
    for (int r = 0; r < 8; r++)
      h1o[(r + 8 * hi) * HP + n0 + nl] = f2bf(fast_tanh(c[r] + bv));
  }
  __syncthreads();

  // ---- layer 2: [16,256] x W2^T -> tanh -> h2, K=256 ----
  for (int t = wid; t < 16; t += 8) {
    int n0 = t << 4;
    v8f c = gemm_tile(h1d, HP, wbf + WOFF_D2, HH, n0, nl, hi);
    float bv = bd2[n0 + nl];
    #pragma unroll
    for (int r = 0; r < 8; r++)
      h2d[(r + 8 * hi) * HP + n0 + nl] = f2bf(fast_tanh(c[r] + bv));
  }
  for (int t = wid; t < 16; t += 8) {
    int n0 = t << 4;
    v8f c = gemm_tile(h1o, HP, wbf + WOFF_O2, HH, n0, nl, hi);
    float bv = bo2[n0 + nl];
    #pragma unroll
    for (int r = 0; r < 8; r++)
      h2o[(r + 8 * hi) * HP + n0 + nl] = f2bf(fast_tanh(c[r] + bv));
  }
  __syncthreads();

  // ---- layer 3: diag (1 tile on waves 4..7) + offdiag (126 tiles, all waves) ----
  if (wid >= 4) {                                     // wave-uniform branch
    int n0 = (wid - 4) << 4;
    v8f c = gemm_tile(h2d, HP, wbf + WOFF_DO, HH, n0, nl, hi);
    float bv = bdo[n0 + nl];
    #pragma unroll
    for (int r = 0; r < 8; r++)
      xd[(r + 8 * hi) * XPF + n0 + nl] = c[r] + bv;
  }
  for (int t = wid; t < 126; t += 8) {
    int n0 = t << 4;
    v8f c = gemm_tile(h2o, HP, wbf + WOFF_OO, HH, n0, nl, hi);
    float bv = boo[n0 + nl];
    #pragma unroll
    for (int r = 0; r < 8; r++)
      zb[(r + 8 * hi) * ZP + n0 + nl] = c[r] + bv;
  }
  __syncthreads();

  // ---- triangular matvecs: 16 threads/sample, 4 rows each ----
  const int s  = tid >> 4;
  const int j0 = tid & 15;
  const float* zr  = zb + s * ZP;
  const float* x0r = x0 + s * XPF;
  const float* xdr = xd + s * XPF;
  float*       yr  = yb + s * XPF;

  // pass 1: y = L^T x0   (y[j] = diag[j]*x0[j] + sum_{k>j} z[k(k-1)/2+j]*x0[k])
  #pragma unroll
  for (int c4 = 0; c4 < 4; c4++) {
    int j = j0 + 16 * c4;
    float acc = xdr[j] * x0r[j];
    int f = (j * (j + 1)) / 2 + j;   // flat(j+1, j)
    int add = j + 1;
    for (int k = j + 1; k < NB; k++) { acc += zr[f] * x0r[k]; f += add; add++; }
    yr[j] = acc;
  }
  __syncthreads();

  // pass 2: D = L y   (D[i] = diag[i]*y[i] + sum_{j<i} z[i(i-1)/2+j]*y[j])
  float* og = out + ((size_t)blockIdx.x * BM + s) * NB;
  #pragma unroll
  for (int c4 = 0; c4 < 4; c4++) {
    int i = j0 + 16 * c4;
    float acc = xdr[i] * yr[i];
    int base = (i * (i - 1)) / 2;
    for (int j = 0; j < i; j++) acc += zr[base + j] * yr[j];
    og[i] = acc;
  }
}

extern "C" void kernel_launch(void* const* d_in, const int* in_sizes, int n_in,
                              void* d_out, int out_size, void* d_ws, size_t ws_size,
                              hipStream_t stream) {
  const float* x   = (const float*)d_in[0];
  const float* Wd1 = (const float*)d_in[1];  const float* bd1 = (const float*)d_in[2];
  const float* Wd2 = (const float*)d_in[3];  const float* bd2 = (const float*)d_in[4];
  const float* Wdo = (const float*)d_in[5];  const float* bdo = (const float*)d_in[6];
  const float* Wo1 = (const float*)d_in[7];  const float* bo1 = (const float*)d_in[8];
  const float* Wo2 = (const float*)d_in[9];  const float* bo2 = (const float*)d_in[10];
  const float* Woo = (const float*)d_in[11]; const float* boo = (const float*)d_in[12];

  ushort_t* wbf = (ushort_t*)d_ws;

  // weight f32 -> bf16 conversions into workspace
  const float* srcs[6] = {Wd1, Wd2, Wdo, Wo1, Wo2, Woo};
  const int    offs[6] = {WOFF_D1, WOFF_D2, WOFF_DO, WOFF_O1, WOFF_O2, WOFF_OO};
  const int    cnts[6] = {HH*NB, HH*HH, NB*HH, HH*NB, HH*HH, OFFD*HH};
  for (int i = 0; i < 6; i++) {
    int blocks = (cnts[i] + 255) / 256;
    wconv_kernel<<<blocks, 256, 0, stream>>>(srcs[i], wbf + offs[i], cnts[i]);
  }

  int B = in_sizes[0] / NB;        // 32768
  int grid = B / BM;               // 2048
  damping_fused_kernel<<<grid, NT, SMEM_TOTAL, stream>>>(
      x, wbf, bd1, bd2, bdo, bo1, bo2, boo, (float*)d_out);
}
